// Attention_27315992002948
// MI455X (gfx1250) — matrix operand order repaired and checked by reading, not compiled, not run on hardware
//
#include <hip/hip_runtime.h>
#include <hip/hip_bf16.h>
#include <math.h>
#include <stdint.h>

// ---------------------------------------------------------------------------
// Multi-head attention for MI455X (gfx1250, wave32, WMMA + TDM).
// Pipeline: cvt(x,W->bf16, W transposed) -> GEMM+bias (Q,K,V, bf16 out)
//           -> per-(b,h,qtile) attention (TDM-staged tiles, WMMA scores,
//              floor/8, softmax, WMMA AV)
//           -> GEMM+bias (output projection, f32 out).
// ---------------------------------------------------------------------------

typedef __bf16 bf16_t;
typedef __attribute__((ext_vector_type(8)))  bf16_t   v8bf;
typedef __attribute__((ext_vector_type(16))) bf16_t   v16bf;
typedef __attribute__((ext_vector_type(8)))  float    v8f;
typedef __attribute__((ext_vector_type(4)))  uint32_t v4u;
typedef __attribute__((ext_vector_type(8)))  uint32_t v8u;

#define WMMA_BF16(a, b, c) \
  __builtin_amdgcn_wmma_f32_16x16x32_bf16(false, (a), false, (b), (short)0, (c), false, false)

constexpr int Bc  = 64;
constexpr int Tc  = 197;
constexpr int Dc  = 768;
constexpr int Hc  = 12;
constexpr int DKc = 64;
constexpr int BT  = Bc * Tc;   // 12608 rows
constexpr int TPAD = 256;      // T padded to multiple of 64 for clean WMMA K-steps

// --- WMMA fragment loaders (wave32 VGPR layouts per CDNA5 ISA 7.12.2) -------
__device__ inline v16bf frag_a_lds(const bf16_t* lds, int stride, int row0, int col0, int lane) {
  int half = (lane >> 4) & 1;
  int m    = lane & 15;
  const bf16_t* p = lds + (row0 + m) * stride + col0 + half * 8;
  v8bf lo = *(const v8bf*)(p);
  v8bf hi = *(const v8bf*)(p + 16);
  return __builtin_shufflevector(lo, hi, 0, 1, 2, 3, 4, 5, 6, 7, 8, 9, 10, 11, 12, 13, 14, 15);
}
__device__ inline v16bf frag_b_lds(const bf16_t* lds, int stride, int n0, int k0, int lane) {
  int half = (lane >> 4) & 1;
  int n    = lane & 15;
  const bf16_t* p = lds + (n0 + n) * stride + k0 + half * 8;
  v8bf lo = *(const v8bf*)(p);
  v8bf hi = *(const v8bf*)(p + 16);
  return __builtin_shufflevector(lo, hi, 0, 1, 2, 3, 4, 5, 6, 7, 8, 9, 10, 11, 12, 13, 14, 15);
}

// --- Tensor Data Mover: 2D tile (tile_rows x 64 bf16, row stride Dc) --------
// D# per CDNA5 ISA 8.3/8.4: count=1, type=2("image"), data_size=2B.
// Rows >= rows_valid are hardware zero-filled (OOB reads return zero).
__device__ inline void tdm_load_2d_64w(const bf16_t* gptr, bf16_t* lptr,
                                       int rows_valid, int tile_rows) {
  uint64_t ga = (uint64_t)(uintptr_t)gptr;
  uint32_t la = (uint32_t)(uintptr_t)lptr;   // LDS aperture: low 32 bits = LDS byte addr
  v4u g0;
  g0[0] = 1u;                                            // count=1, user descriptor
  g0[1] = la;                                            // lds_addr
  g0[2] = (uint32_t)ga;                                  // global_addr[31:0]
  g0[3] = (uint32_t)((ga >> 32) & 0x01FFFFFFu) | 0x80000000u;  // addr[56:32] | type=2
  uint32_t td1 = (uint32_t)rows_valid;
  v8u g1;
  g1[0] = 0x00010000u;                                   // wg_mask=0, data_size=1 (2 bytes)
  g1[1] = (64u << 16);                                   // tensor_dim0 = 64 (bits 63:48)
  g1[2] = (td1 & 0xFFFFu) << 16;                         // tensor_dim1 lo (bits 95:80)
  g1[3] = (td1 >> 16) | (64u << 16);                     // tensor_dim1 hi | tile_dim0=64
  g1[4] = (uint32_t)tile_rows;                           // tile_dim1 | tile_dim2=0
  g1[5] = (uint32_t)Dc;                                  // tensor_dim0_stride = 768
  g1[6] = 0u;
  g1[7] = 0u;
  asm volatile("tensor_load_to_lds %0, %1" : : "s"(g0), "s"(g1) : "memory");
}

// --- conversion kernels -----------------------------------------------------
__global__ __launch_bounds__(256) void cvt_f32_bf16_kernel(const float* __restrict__ in,
                                                           bf16_t* __restrict__ out, int n) {
  int i = blockIdx.x * 256 + threadIdx.x;
  if (i < n) out[i] = (bf16_t)in[i];
}

// W [K,N] f32 -> Wt [N,K] bf16 (transposed so B-fragments are contiguous-K loads)
__global__ __launch_bounds__(256) void transpose_f32_bf16_kernel(const float* __restrict__ W,
                                                                 bf16_t* __restrict__ Wt, int d) {
  int i = blockIdx.x * 256 + threadIdx.x;
  if (i < d * d) {
    int k = i / d, n = i % d;
    Wt[(size_t)n * d + k] = (bf16_t)W[i];
  }
}

// --- GEMM + bias: C[M,N] = A[M,K] @ Bt[N,K]^T + bias ------------------------
// 256 threads = 8 wave32; block tile 128(M)x64(N); K-step 32.
// Each wave: one 16-row M band, 4 N-frags sharing one A-frag (4 WMMAs/K-step).
template <typename OutT>
__global__ __launch_bounds__(256) void gemm_bias_kernel(const bf16_t* __restrict__ A,
                                                        const bf16_t* __restrict__ Bt,
                                                        const float* __restrict__ bias,
                                                        OutT* __restrict__ C,
                                                        int M, int N, int K) {
  __shared__ __attribute__((aligned(16))) bf16_t sA[128 * 32];
  __shared__ __attribute__((aligned(16))) bf16_t sB[64 * 32];

  const int tid  = threadIdx.x;
  const int lane = tid & 31;
  const int w    = tid >> 5;
  const int m0 = blockIdx.x * 128;
  const int n0 = blockIdx.y * 64;
  const int mBand = w * 16;          // 8 waves -> 128 M rows

  v8f acc[4] = {};

  const int rB = tid >> 2;           // 0..63
  const int cB = (tid & 3) * 8;      // 0..24

  for (int kk = 0; kk < K; kk += 32) {
#pragma unroll
    for (int rep = 0; rep < 2; ++rep) {     // stage A: 128x32
      int idx = tid + rep * 256;
      int rA = idx >> 2, cA = (idx & 3) * 8;
      v8bf av = {};
      if (m0 + rA < M) av = *(const v8bf*)&A[(size_t)(m0 + rA) * K + kk + cA];
      *(v8bf*)&sA[rA * 32 + cA] = av;
    }
    {                                        // stage B: 64x32
      v8bf bv = {};
      if (n0 + rB < N) bv = *(const v8bf*)&Bt[(size_t)(n0 + rB) * K + kk + cB];
      *(v8bf*)&sB[rB * 32 + cB] = bv;
    }
    if (kk + 32 < K) {  // global_prefetch of the next K-slab
      if (m0 + rB * 2 < M) __builtin_prefetch(&A[(size_t)(m0 + rB * 2) * K + kk + 32 + cB], 0, 3);
      if (n0 + rB < N)     __builtin_prefetch(&Bt[(size_t)(n0 + rB) * K + kk + 32 + cB], 0, 3);
    }
    __syncthreads();
    v16bf a = frag_a_lds(sA, 32, mBand, 0, lane);
#pragma unroll
    for (int f = 0; f < 4; ++f) {
      v16bf b = frag_b_lds(sB, 32, f * 16, 0, lane);
      acc[f] = WMMA_BF16(a, b, acc[f]);
    }
    __syncthreads();
  }

  const int half = (lane >> 4) & 1;
#pragma unroll
  for (int f = 0; f < 4; ++f) {
    int n = n0 + f * 16 + (lane & 15);
    float bb = bias[n];
#pragma unroll
    for (int j = 0; j < 8; ++j) {
      int m = m0 + mBand + half * 8 + j;
      if (m < M) C[(size_t)m * N + n] = (OutT)(acc[f][j] + bb);
    }
  }
}

// --- attention: one block per (qtile of 32, h, b) ---------------------------
__global__ __launch_bounds__(256) void attn_kernel(const bf16_t* __restrict__ Qb,
                                                   const bf16_t* __restrict__ Kb,
                                                   const bf16_t* __restrict__ Vb,
                                                   bf16_t* __restrict__ Cb) {
  __shared__ __attribute__((aligned(16))) float  sS[32 * TPAD];   // scores (f32)
  __shared__ __attribute__((aligned(16))) bf16_t sP[32 * TPAD];   // softmax probs (bf16)
  __shared__ __attribute__((aligned(16))) bf16_t sQ[32 * DKc];    // Q tile
  __shared__ __attribute__((aligned(16))) bf16_t sKV[64 * 64];    // K chunk / V^T chunk

  const int tid  = threadIdx.x;
  const int lane = tid & 31;
  const int w    = tid >> 5;
  const int q0   = blockIdx.x * 32;
  const int h    = blockIdx.y;
  const int bIdx = blockIdx.z;

  const int qhalf = w & 1;         // which 16-row query half
  const int sub   = (w >> 1) & 3;  // which 16-wide tile within the 64-chunk

  // ---- stage Q tile (32 x 64) via TDM; rows >= T hardware zero-filled ----
  if (w == 0) {
    tdm_load_2d_64w(&Qb[(size_t)(bIdx * Tc + q0) * Dc + h * DKc], sQ, Tc - q0, 32);
  }
  __builtin_amdgcn_s_wait_tensorcnt(0);
  __syncthreads();

  const v16bf qa0 = frag_a_lds(sQ, DKc, qhalf * 16, 0, lane);
  const v16bf qa1 = frag_a_lds(sQ, DKc, qhalf * 16, 32, lane);

  // ---- phase 1: scores S = floor((Q K^T) / 8), mask cols >= T with -inf ----
  for (int c = 0; c < TPAD / 64; ++c) {
    // stage K chunk (64 x 64) via TDM, natural [t][dk] layout, zero-padded
    if (w == 0) {
      tdm_load_2d_64w(&Kb[(size_t)(bIdx * Tc + c * 64) * Dc + h * DKc], sKV, Tc - c * 64, 64);
    }
    __builtin_amdgcn_s_wait_tensorcnt(0);
    __syncthreads();

    const int kt = c * 4 + sub;  // global 16-wide key tile index, 0..15
    v8f s = {};
    v16bf b0 = frag_b_lds(sKV, 64, sub * 16, 0, lane);
    v16bf b1 = frag_b_lds(sKV, 64, sub * 16, 32, lane);
    s = WMMA_BF16(qa0, b0, s);
    s = WMMA_BF16(qa1, b1, s);

    const int half = (lane >> 4) & 1;
    const int col  = kt * 16 + (lane & 15);
#pragma unroll
    for (int j = 0; j < 8; ++j) {
      int row = qhalf * 16 + half * 8 + j;
      float sv = floorf(s[j] * 0.125f);  // floor(scores / sqrt(64))
      if (col >= Tc) sv = -INFINITY;
      sS[row * TPAD + col] = sv;
    }
    __syncthreads();
  }

  // ---- phase 2: row softmax (one thread per query row) ----
  if (tid < 32) {
    const float* Srow = &sS[tid * TPAD];
    bf16_t* Prow = &sP[tid * TPAD];
    float m = -INFINITY;
    for (int k = 0; k < TPAD; ++k) m = fmaxf(m, Srow[k]);
    float sum = 0.f;
    for (int k = 0; k < TPAD; ++k) {
      float e = __expf(Srow[k] - m);
      sum += e;
      Prow[k] = (bf16_t)e;
    }
    float inv = 1.f / sum;
    for (int k = 0; k < TPAD; ++k) Prow[k] = (bf16_t)((float)Prow[k] * inv);
  }
  __syncthreads();

  // ---- phase 3: context = P @ V ----
  const int dkt = sub;  // 16-wide dk tile, 0..3
  v8f acc = {};
  for (int c = 0; c < TPAD / 64; ++c) {
    // stage V^T chunk: sKV[dk][tl] = V[c*64+tl][dk], zero-padded (manual: transpose)
#pragma unroll
    for (int rep = 0; rep < 16; ++rep) {
      int idx = tid + rep * 256;
      int tl = idx & 63;
      int dk = idx >> 6;
      int t = c * 64 + tl;
      bf16_t v = (bf16_t)0.0f;
      if (t < Tc) v = Vb[(size_t)(bIdx * Tc + t) * Dc + h * DKc + dk];
      sKV[dk * 64 + tl] = v;
    }
    __syncthreads();
#pragma unroll
    for (int kk = 0; kk < 64; kk += 32) {
      v16bf a = frag_a_lds(sP, TPAD, qhalf * 16, c * 64 + kk, lane);
      v16bf b = frag_b_lds(sKV, 64, dkt * 16, kk, lane);
      acc = WMMA_BF16(a, b, acc);
    }
    __syncthreads();
  }

  // ---- store context tile (bf16) ----
  const int half = (lane >> 4) & 1;
  const int nIdx = dkt * 16 + (lane & 15);
#pragma unroll
  for (int j = 0; j < 8; ++j) {
    int q = q0 + qhalf * 16 + half * 8 + j;
    if (q < Tc) Cb[(size_t)(bIdx * Tc + q) * Dc + h * DKc + nIdx] = (bf16_t)acc[j];
  }
}

// ---------------------------------------------------------------------------
extern "C" void kernel_launch(void* const* d_in, const int* in_sizes, int n_in,
                              void* d_out, int out_size, void* d_ws, size_t ws_size,
                              hipStream_t stream) {
  const float* x  = (const float*)d_in[0];
  const float* Wq = (const float*)d_in[1];
  const float* bq = (const float*)d_in[2];
  const float* Wk = (const float*)d_in[3];
  const float* bk = (const float*)d_in[4];
  const float* Wv = (const float*)d_in[5];
  const float* bv = (const float*)d_in[6];
  const float* Wo = (const float*)d_in[7];
  const float* bo = (const float*)d_in[8];

  // carve workspace (all bf16 staging buffers), 256B-aligned slices
  char* p = (char*)d_ws;
  auto carve = [&](size_t bytes) {
    void* r = (void*)p;
    p += (bytes + 255) & ~(size_t)255;
    return r;
  };
  bf16_t* xb  = (bf16_t*)carve((size_t)BT * Dc * sizeof(bf16_t));
  bf16_t* Wqt = (bf16_t*)carve((size_t)Dc * Dc * sizeof(bf16_t));
  bf16_t* Wkt = (bf16_t*)carve((size_t)Dc * Dc * sizeof(bf16_t));
  bf16_t* Wvt = (bf16_t*)carve((size_t)Dc * Dc * sizeof(bf16_t));
  bf16_t* Wot = (bf16_t*)carve((size_t)Dc * Dc * sizeof(bf16_t));
  bf16_t* Qb  = (bf16_t*)carve((size_t)BT * Dc * sizeof(bf16_t));
  bf16_t* Kb  = (bf16_t*)carve((size_t)BT * Dc * sizeof(bf16_t));
  bf16_t* Vb  = (bf16_t*)carve((size_t)BT * Dc * sizeof(bf16_t));
  bf16_t* Cb  = (bf16_t*)carve((size_t)BT * Dc * sizeof(bf16_t));

  // 1) precision conversion + weight transposes
  {
    int n = BT * Dc;
    cvt_f32_bf16_kernel<<<(n + 255) / 256, 256, 0, stream>>>(x, xb, n);
    int nw = Dc * Dc;
    int gw = (nw + 255) / 256;
    transpose_f32_bf16_kernel<<<gw, 256, 0, stream>>>(Wq, Wqt, Dc);
    transpose_f32_bf16_kernel<<<gw, 256, 0, stream>>>(Wk, Wkt, Dc);
    transpose_f32_bf16_kernel<<<gw, 256, 0, stream>>>(Wv, Wvt, Dc);
    transpose_f32_bf16_kernel<<<gw, 256, 0, stream>>>(Wo, Wot, Dc);
  }

  // 2) Q/K/V projections (bf16 out, f32 accumulate)
  dim3 gGemm((BT + 127) / 128, Dc / 64);  // 99 x 12
  gemm_bias_kernel<bf16_t><<<gGemm, 256, 0, stream>>>(xb, Wqt, bq, Qb, BT, Dc, Dc);
  gemm_bias_kernel<bf16_t><<<gGemm, 256, 0, stream>>>(xb, Wkt, bk, Kb, BT, Dc, Dc);
  gemm_bias_kernel<bf16_t><<<gGemm, 256, 0, stream>>>(xb, Wvt, bv, Vb, BT, Dc, Dc);

  // 3) attention per (b, h, 32-query tile)
  dim3 gAttn((Tc + 31) / 32, Hc, Bc);  // 7 x 12 x 64
  attn_kernel<<<gAttn, 256, 0, stream>>>(Qb, Kb, Vb, Cb);

  // 4) output projection (f32 out)
  gemm_bias_kernel<float><<<gGemm, 256, 0, stream>>>(Cb, Wot, bo, (float*)d_out, BT, Dc, Dc);
}
